// Yolact_24713241821404
// MI455X (gfx1250) — matrix operand order repaired and checked by reading, not compiled, not run on hardware
//
#include <hip/hip_runtime.h>
#include <hip/hip_bf16.h>

typedef __attribute__((ext_vector_type(16))) _Float16 v16h;
typedef __attribute__((ext_vector_type(8)))  float    v8f;
typedef __attribute__((ext_vector_type(4)))  unsigned int u32x4;
typedef __attribute__((ext_vector_type(8)))  int      i32x8;
typedef __attribute__((ext_vector_type(4)))  int      i32x4;

#define NA_TOTAL 19248
#define TOPK 100
#define MASK_DIM 32
#define NCLS 81

// ---------------------------------------------------------------------------
// Tensor Data Mover: load a tile_h x tile_w tile of a 2-D f16 tensor
// (row stride = tensor_w elements) into LDS at byte offset ldsOff.
// D# per CDNA5 ISA. 2-D tensor -> groups 2/3 zero.
// ---------------------------------------------------------------------------
__device__ inline void tdm_load_2d_f16(const _Float16* src, unsigned ldsOff,
                                       int tensor_w, int tensor_h,
                                       int tile_w, int tile_h)
{
    unsigned long long ga = (unsigned long long)(size_t)src;
    u32x4 g0 = {0u, 0u, 0u, 0u};
    g0[0] = 1u;                                        // count=1, user mode
    g0[1] = ldsOff;                                    // lds_addr (bytes)
    g0[2] = (unsigned)(ga & 0xffffffffu);              // global_addr[31:0]
    g0[3] = (unsigned)((ga >> 32) & 0x01ffffffu)       // global_addr[56:32]
          | (2u << 30);                                // type = 2 ("image")
    unsigned long long s0 = (unsigned long long)(unsigned)tensor_w; // dim0 stride (elems)
    i32x8 g1 = {0, 0, 0, 0, 0, 0, 0, 0};
    g1[0] = (int)(1u << 16);                           // data_size = 1 -> 2 bytes
    g1[1] = (int)(((unsigned)tensor_w & 0xffffu) << 16);           // tensor_dim0 lo16
    g1[2] = (int)((((unsigned)tensor_w >> 16) & 0xffffu)
          | (((unsigned)tensor_h & 0xffffu) << 16));               // dim0 hi / dim1 lo
    g1[3] = (int)((((unsigned)tensor_h >> 16) & 0xffffu)
          | (((unsigned)tile_w & 0xffffu) << 16));                 // dim1 hi / tile_dim0
    g1[4] = (int)((unsigned)tile_h & 0xffffu);                     // tile_dim1 (tile_dim2=0)
    g1[5] = (int)(unsigned)(s0 & 0xffffffffu);                     // dim0_stride lo32
    g1[6] = (int)(unsigned)((s0 >> 32) & 0xffffu);                 // dim0_stride hi16
    i32x4 z4 = {0, 0, 0, 0};
#if defined(__clang_major__) && (__clang_major__ >= 23)
    i32x8 z8 = {0, 0, 0, 0, 0, 0, 0, 0};
    __builtin_amdgcn_tensor_load_to_lds(g0, g1, z4, z4, z8, 0);
#else
    __builtin_amdgcn_tensor_load_to_lds(g0, g1, z4, z4, 0);
#endif
}

// ---------------------------------------------------------------------------
// Implicit-GEMM convolution on CDNA5 WMMA.
//   D[M=Cout, N=B*Hout*Wout] = W16[M,K] * im2col(X)[K,N] + bias
//   K = Cin*KH*KW (multiple of 32 for every conv in this network).
// 256 threads = 8 waves, block tile 64(M) x 128(N), K-step 32.
// Waves: 4(M) x 2(N); each wave -> 4 x v_wmma_f32_16x16x32_f16 per step.
// A tile staged in LDS by the Tensor Data Mover; B tile by cooperative
// branchless im2col gather (clamped address + post-select, no EXEC
// branching). act: 0 none, 1 relu, 2 tanh.
// ---------------------------------------------------------------------------
template <int KH, int KW>
__global__ __launch_bounds__(256)
void conv_wmma(const float* __restrict__ in, const _Float16* __restrict__ w16,
               const float* __restrict__ bias, float* __restrict__ out,
               int B, int Cin, int Hin, int Win,
               int Cout, int Hout, int Wout,
               int stride, int pad, int act)
{
    __shared__ _Float16 As[64 * 32];    // [m][k]   4 KB
    __shared__ _Float16 Bs[128 * 32];   // [n][k]   8 KB

    const int tid    = threadIdx.x;
    const int lane   = tid & 31;
    const int wave   = tid >> 5;        // 0..7
    const int wave_m = wave >> 1;       // 0..3  -> M offset 16*wave_m
    const int wave_n = wave & 1;        // 0..1  -> N offset 64*wave_n
    const int laneHi = (lane >> 4) & 1;
    const int m      = lane & 15;

    const int K    = Cin * KH * KW;
    const int HW   = Hout * Wout;
    const int HWin = Hin * Win;
    const int Ntot = B * HW;
    const int co_blk = blockIdx.x * 64;
    const int n_blk  = blockIdx.y * 128;

    // ---- B-tile loader mapping: thread -> (pixel column, 16 K values) ----
    const int ln  = tid & 127;          // n_local
    const int lkh = (tid >> 7) * 16;    // k-half: 0 or 16
    int lb, lyy, lxx; bool lvalid;
    {
        int n = n_blk + ln;
        lvalid = (n < Ntot);
        int nn = lvalid ? n : 0;
        lb = nn / HW; int r = nn - lb * HW;
        lyy = r / Wout; lxx = r - lyy * Wout;
    }
    const float* inb = in + (size_t)lb * Cin * HWin;

    // ---- per-wave output-tile pixel decode ----
    int tb[4], toy[4], tox[4]; bool tval[4];
#pragma unroll
    for (int t = 0; t < 4; ++t) {
        int n = n_blk + wave_n * 64 + 16 * t + m;
        tval[t] = (n < Ntot);
        int nn = tval[t] ? n : 0;
        tb[t] = nn / HW; int r = nn - tb[t] * HW;
        toy[t] = r / Wout; tox[t] = r - toy[t] * Wout;
    }

    // ---- C init = bias, branchless (clamped read + select) ----
    v8f acc[4];
    {
        int co = co_blk + wave_m * 16 + (laneHi ? 8 : 0);
#pragma unroll
        for (int v = 0; v < 8; ++v) {
            int cov = co + v;
            float bv = bias[min(cov, Cout - 1)];
            float b0 = (cov < Cout) ? bv : 0.f;
            acc[0][v] = b0;
        }
    }
#pragma unroll
    for (int t = 1; t < 4; ++t) acc[t] = acc[0];

    const unsigned ldsA = (unsigned)(size_t)(&As[0]);

    for (int ks = 0; ks < K; ks += 32) {
        __syncthreads();   // previous fragments consumed before restage

        // A tile: rows [co_blk, co_blk+64) x cols [ks, ks+32) of W16[Cout][K].
        // TDM OOB rows are zero-filled; epilogue masks co >= Cout anyway.
        if (wave == 0)
            tdm_load_2d_f16(w16 + (size_t)co_blk * K + ks, ldsA, K, Cout, 32, 64);

        // B tile: branchless im2col gather (clamp address, select 0 after)
        {
            int kk = ks + lkh;
#pragma unroll
            for (int i = 0; i < 16; ++i) {
                int k = kk + i;
                int ci, ky, kx;
                if (KH * KW == 1) { ci = k; ky = 0; kx = 0; }
                else {
                    ci = k / (KH * KW);
                    int r = k - ci * (KH * KW);
                    ky = r / KW;
                    kx = r - ky * KW;
                }
                int iy = lyy * stride + ky - pad;
                int ix = lxx * stride + kx - pad;
                bool ok = lvalid & ((unsigned)iy < (unsigned)Hin)
                                 & ((unsigned)ix < (unsigned)Win);
                int iyc = min(max(iy, 0), Hin - 1);
                int ixc = min(max(ix, 0), Win - 1);
                float val = inb[(size_t)ci * HWin + iyc * Win + ixc];
                Bs[ln * 32 + lkh + i] = (_Float16)(ok ? val : 0.f);
            }
        }

        if (wave == 0) __builtin_amdgcn_s_wait_tensorcnt(0);
        __syncthreads();   // A (TDM) + B (ds stores) visible to all waves

        // ---- fragments from LDS + WMMA ----
        // A 16x32: lane m&15 = row; VGPR v -> K pair {(v<4?2v:2v+8)+8*laneHi, +1}
        v16h a;
        {
            const _Float16* ap = &As[(wave_m * 16 + m) * 32];
#pragma unroll
            for (int v = 0; v < 8; ++v) {
                int kb = ((v < 4) ? 2 * v : 2 * v + 8) + (laneHi ? 8 : 0);
                a[2 * v]     = ap[kb];
                a[2 * v + 1] = ap[kb + 1];
            }
        }
        // B 32x16: col = lane&15; VGPR v -> K pair {(v<4?2v:2v+8)+8*laneHi, +1}
#pragma unroll
        for (int t = 0; t < 4; ++t) {
            v16h bf;
            const _Float16* bp = &Bs[(wave_n * 64 + 16 * t + m) * 32];
#pragma unroll
            for (int v = 0; v < 8; ++v) {
                int kb = ((v < 4) ? 2 * v : 2 * v + 8) + (laneHi ? 8 : 0);
                bf[2 * v]     = bp[kb];
                bf[2 * v + 1] = bp[kb + 1];
            }
            acc[t] = __builtin_amdgcn_wmma_f32_16x16x32_f16(
                false, a, false, bf, (short)0, acc[t], false, false);
        }
    }

    // ---- epilogue: activation + NCHW store (predication required here) ----
#pragma unroll
    for (int t = 0; t < 4; ++t) {
        if (!tval[t]) continue;
#pragma unroll
        for (int v = 0; v < 8; ++v) {
            int co = co_blk + wave_m * 16 + v + (laneHi ? 8 : 0);
            if (co >= Cout) continue;
            float r = acc[t][v];
            if (act == 1)      r = fmaxf(r, 0.f);
            else if (act == 2) r = tanhf(r);
            out[(((size_t)tb[t] * Cout + co) * Hout + toy[t]) * Wout + tox[t]] = r;
        }
    }
}

// ---------------------------------------------------------------------------
__global__ void f32_to_f16(const float* __restrict__ src, _Float16* __restrict__ dst, int n)
{
    int i = blockIdx.x * blockDim.x + threadIdx.x;
    if (i < n) dst[i] = (_Float16)src[i];
}

// ---------------------------------------------------------------------------
// half-pixel bilinear resize (jax.image.resize 'bilinear'), optional +=
// ---------------------------------------------------------------------------
__global__ void upsample_bilinear(const float* __restrict__ src, float* __restrict__ dst,
                                  int BC, int Hs, int Ws, int Hd, int Wd, int add)
{
    int idx = blockIdx.x * blockDim.x + threadIdx.x;
    int total = BC * Hd * Wd;
    if (idx >= total) return;
    int x = idx % Wd; int t = idx / Wd;
    int y = t % Hd;   int bc = t / Hd;
    float fy = (y + 0.5f) * (float)Hs / (float)Hd - 0.5f;
    float fx = (x + 0.5f) * (float)Ws / (float)Wd - 0.5f;
    int y0 = (int)floorf(fy); float wy = fy - y0;
    int x0 = (int)floorf(fx); float wx = fx - x0;
    int y1 = y0 + 1, x1 = x0 + 1;
    y0 = max(0, min(Hs - 1, y0)); y1 = max(0, min(Hs - 1, y1));
    x0 = max(0, min(Ws - 1, x0)); x1 = max(0, min(Ws - 1, x1));
    const float* sp = src + (size_t)bc * Hs * Ws;
    float v = (1.f - wy) * ((1.f - wx) * sp[y0 * Ws + x0] + wx * sp[y0 * Ws + x1])
            +        wy  * ((1.f - wx) * sp[y1 * Ws + x0] + wx * sp[y1 * Ws + x1]);
    if (add) dst[idx] += v; else dst[idx] = v;
}

// ---------------------------------------------------------------------------
// head output [B, 3*Cp, H, W] -> anchor layout dst[B, NA_TOTAL, Cp] slice
// ---------------------------------------------------------------------------
__global__ void scatter_head(const float* __restrict__ src, float* __restrict__ dst,
                             int H, int W, int Cp, int base)
{
    long long idx = (long long)blockIdx.x * blockDim.x + threadIdx.x;
    long long total = 4LL * H * W * 3 * Cp;
    if (idx >= total) return;
    int c = (int)(idx % Cp); long long t = idx / Cp;
    int a = (int)(t % 3);  t /= 3;
    int x = (int)(t % W);  t /= W;
    int y = (int)(t % H);  t /= H;
    int b = (int)t;
    int ch = a * Cp + c;
    dst[((size_t)b * NA_TOTAL + base + (y * W + x) * 3 + a) * Cp + c] =
        src[(((size_t)b * (3 * Cp) + ch) * H + y) * W + x];
}

// ---------------------------------------------------------------------------
// analytic priors + box decode + softmax best-nonbg score (into topk scratch)
// ---------------------------------------------------------------------------
__global__ void decode_score(const float* __restrict__ loc, const float* __restrict__ conf,
                             float* __restrict__ boxes, float* __restrict__ scratch)
{
    int i = blockIdx.x * blockDim.x + threadIdx.x;
    if (i >= 4 * NA_TOTAL) return;
    int j = i % NA_TOTAL;
    const int   loff[6] = {0, 14283, 17958, 18930, 19173, 19248};
    const int   lf[5]   = {69, 35, 18, 9, 5};
    const float lsc[5]  = {24.f, 48.f, 96.f, 192.f, 384.f};
    const float sar[3]  = {1.f, 0.70710678f, 1.41421356f};
    int L = 0;
    while (L < 4 && j >= loff[L + 1]) ++L;
    int jl = j - loff[L];
    int cell = jl / 3, a = jl % 3;
    int f = lf[L];
    float cx = ((cell % f) + 0.5f) / (float)f;
    float cy = ((cell / f) + 0.5f) / (float)f;
    float pw = lsc[L] * sar[a] / 550.f;
    float ph = lsc[L] / (sar[a] * 550.f);
    const float* lp = loc + (size_t)i * 4;
    float ncx = cx + lp[0] * 0.1f * pw;
    float ncy = cy + lp[1] * 0.1f * ph;
    float nw  = pw * expf(lp[2] * 0.2f);
    float nh  = ph * expf(lp[3] * 0.2f);
    float* bp = boxes + (size_t)i * 4;
    bp[0] = ncx - nw * 0.5f; bp[1] = ncy - nh * 0.5f;
    bp[2] = ncx + nw * 0.5f; bp[3] = ncy + nh * 0.5f;
    const float* cp = conf + (size_t)i * NCLS;
    float mx = -1e30f;
    for (int c = 0; c < NCLS; ++c) mx = fmaxf(mx, cp[c]);
    float den = 0.f, best = -1e30f;
    for (int c = 0; c < NCLS; ++c) {
        float e = expf(cp[c] - mx);
        den += e;
        if (c >= 1) best = fmaxf(best, e);
    }
    scratch[i] = best / den;
}

// ---------------------------------------------------------------------------
// top-100 per batch: iterative argmax with LDS reduction; mutates scratch
// (scratch is rewritten by decode_score every launch -> deterministic)
// ---------------------------------------------------------------------------
__global__ void topk_kernel(float* __restrict__ scratch, int* __restrict__ topidx,
                            float* __restrict__ top_s)
{
    __shared__ float sv[256];
    __shared__ int   si[256];
    int b = blockIdx.x, tid = threadIdx.x;
    float* sc = scratch + (size_t)b * NA_TOTAL;
    for (int k = 0; k < TOPK; ++k) {
        float best = -1e30f; int bi = 0x7fffffff;
        for (int i = tid; i < NA_TOTAL; i += blockDim.x) {
            float v = sc[i];
            if (v > best || (v == best && i < bi)) { best = v; bi = i; }
        }
        sv[tid] = best; si[tid] = bi;
        __syncthreads();
        for (int s = 128; s > 0; s >>= 1) {
            if (tid < s) {
                if (sv[tid + s] > sv[tid] ||
                    (sv[tid + s] == sv[tid] && si[tid + s] < si[tid])) {
                    sv[tid] = sv[tid + s]; si[tid] = si[tid + s];
                }
            }
            __syncthreads();
        }
        if (tid == 0) {
            topidx[b * TOPK + k] = si[0];
            top_s [b * TOPK + k] = sv[0];
            sc[si[0]] = -1e38f;
        }
        __syncthreads();
    }
}

__global__ void gather_top(const float* __restrict__ boxes, const float* __restrict__ coef,
                           const int* __restrict__ topidx,
                           float* __restrict__ out_boxes, float* __restrict__ ctop)
{
    int i = blockIdx.x * blockDim.x + threadIdx.x;
    if (i >= 4 * TOPK) return;
    int b = i / TOPK;
    int idx = topidx[i];
    for (int c = 0; c < 4; ++c)
        out_boxes[i * 4 + c] = boxes[((size_t)b * NA_TOTAL + idx) * 4 + c];
    for (int m = 0; m < MASK_DIM; ++m)
        ctop[i * MASK_DIM + m] = coef[((size_t)b * NA_TOTAL + idx) * MASK_DIM + m];
}

// ---------------------------------------------------------------------------
// masks[b,y,x,k] = sigmoid( sum_m proto[b,m,y,x] * ctop[b,k,m] )
// ---------------------------------------------------------------------------
__global__ void mask_kernel(const float* __restrict__ proto, const float* __restrict__ ctop,
                            float* __restrict__ out, int HW)
{
    __shared__ float sc[TOPK * MASK_DIM];
    int b = blockIdx.y;
    for (int i = threadIdx.x; i < TOPK * MASK_DIM; i += blockDim.x)
        sc[i] = ctop[(size_t)b * TOPK * MASK_DIM + i];
    __syncthreads();
    int p = blockIdx.x * blockDim.x + threadIdx.x;
    if (p >= HW) return;
    float pr[MASK_DIM];
    const float* pp = proto + (size_t)b * MASK_DIM * HW + p;
#pragma unroll
    for (int m = 0; m < MASK_DIM; ++m) pr[m] = pp[(size_t)m * HW];
    float* op = out + ((size_t)b * HW + p) * TOPK;
    for (int k = 0; k < TOPK; ++k) {
        float s = 0.f;
#pragma unroll
        for (int m = 0; m < MASK_DIM; ++m) s += pr[m] * sc[k * MASK_DIM + m];
        op[k] = 1.f / (1.f + expf(-s));
    }
}

// ---------------------------------------------------------------------------
// host orchestration
// ---------------------------------------------------------------------------
static void launch_conv(hipStream_t s, const float* in, const _Float16* w16, const float* bias,
                        float* out, int B, int Cin, int Hin, int Win,
                        int Cout, int Hout, int Wout, int stride, int pad, int act, bool k3)
{
    int N = B * Hout * Wout;
    dim3 grid((Cout + 63) / 64, (N + 127) / 128);
    if (k3)
        conv_wmma<3, 3><<<grid, 256, 0, s>>>(in, w16, bias, out, B, Cin, Hin, Win,
                                             Cout, Hout, Wout, stride, pad, act);
    else
        conv_wmma<1, 1><<<grid, 256, 0, s>>>(in, w16, bias, out, B, Cin, Hin, Win,
                                             Cout, Hout, Wout, stride, pad, act);
}

extern "C" void kernel_launch(void* const* d_in, const int* in_sizes, int n_in,
                              void* d_out, int out_size, void* d_ws, size_t ws_size,
                              hipStream_t stream)
{
    (void)in_sizes; (void)n_in; (void)out_size; (void)ws_size;
    const float* c3 = (const float*)d_in[0];
    const float* c4 = (const float*)d_in[1];
    const float* c5 = (const float*)d_in[2];
    const float* lat3_w = (const float*)d_in[3];  const float* lat3_b = (const float*)d_in[4];
    const float* lat4_w = (const float*)d_in[5];  const float* lat4_b = (const float*)d_in[6];
    const float* lat5_w = (const float*)d_in[7];  const float* lat5_b = (const float*)d_in[8];
    const float* fpn3_w = (const float*)d_in[9];  const float* fpn3_b = (const float*)d_in[10];
    const float* fpn4_w = (const float*)d_in[11]; const float* fpn4_b = (const float*)d_in[12];
    const float* fpn5_w = (const float*)d_in[13]; const float* fpn5_b = (const float*)d_in[14];
    const float* ds1_w  = (const float*)d_in[15]; const float* ds1_b  = (const float*)d_in[16];
    const float* ds2_w  = (const float*)d_in[17]; const float* ds2_b  = (const float*)d_in[18];
    const float* up_w   = (const float*)d_in[19]; const float* up_b   = (const float*)d_in[20];
    const float* bbox_w = (const float*)d_in[21]; const float* bbox_b = (const float*)d_in[22];
    const float* conf_w = (const float*)d_in[23]; const float* conf_b = (const float*)d_in[24];
    const float* mask_w = (const float*)d_in[25]; const float* mask_b = (const float*)d_in[26];
    const float* pr1_w  = (const float*)d_in[27]; const float* pr1_b  = (const float*)d_in[28];
    const float* pr2_w  = (const float*)d_in[29]; const float* pr2_b  = (const float*)d_in[30];
    const float* pr3_w  = (const float*)d_in[31]; const float* pr3_b  = (const float*)d_in[32];
    const float* pr4_w  = (const float*)d_in[33]; const float* pr4_b  = (const float*)d_in[34];
    const float* pr5_w  = (const float*)d_in[35]; const float* pr5_b  = (const float*)d_in[36];

    const int B = 4, F = 256;

    // ---- workspace layout, 256B aligned ----
    char* ws = (char*)d_ws;
    size_t off = 0;
    auto allocf = [&](size_t nfloats) -> float* {
        float* p = (float*)(ws + off);
        off += ((nfloats * sizeof(float) + 255) / 256) * 256;
        return p;
    };
    auto alloch = [&](size_t nhalf) -> _Float16* {
        _Float16* p = (_Float16*)(ws + off);
        off += ((nhalf * sizeof(_Float16) + 255) / 256) * 256;
        return p;
    };

    // f16 weight repack targets
    const int K33 = 256 * 9;
    _Float16* w16_lat3 = alloch((size_t)F * 512);
    _Float16* w16_lat4 = alloch((size_t)F * 1024);
    _Float16* w16_lat5 = alloch((size_t)F * 2048);
    _Float16* w16_fpn3 = alloch((size_t)F * K33);
    _Float16* w16_fpn4 = alloch((size_t)F * K33);
    _Float16* w16_fpn5 = alloch((size_t)F * K33);
    _Float16* w16_ds1  = alloch((size_t)F * K33);
    _Float16* w16_ds2  = alloch((size_t)F * K33);
    _Float16* w16_up   = alloch((size_t)F * K33);
    _Float16* w16_bbox = alloch((size_t)12 * K33);
    _Float16* w16_conf = alloch((size_t)243 * K33);
    _Float16* w16_mask = alloch((size_t)96 * K33);
    _Float16* w16_pr1  = alloch((size_t)F * K33);
    _Float16* w16_pr2  = alloch((size_t)F * K33);
    _Float16* w16_pr3  = alloch((size_t)F * K33);
    _Float16* w16_pr4  = alloch((size_t)F * K33);
    _Float16* w16_pr5  = alloch((size_t)MASK_DIM * F);

    float* x3    = allocf((size_t)B * F * 69 * 69);
    float* x4    = allocf((size_t)B * F * 35 * 35);
    float* x5    = allocf((size_t)B * F * 18 * 18);
    float* p3    = allocf((size_t)B * F * 69 * 69);
    float* p4    = allocf((size_t)B * F * 35 * 35);
    float* p5    = allocf((size_t)B * F * 18 * 18);
    float* p6    = allocf((size_t)B * F * 9 * 9);
    float* p7    = allocf((size_t)B * F * 5 * 5);
    float* qa    = allocf((size_t)B * F * 138 * 138);
    float* qb    = allocf((size_t)B * F * 138 * 138);
    float* proto = allocf((size_t)B * MASK_DIM * 138 * 138);
    float* th    = allocf((size_t)B * F * 69 * 69);
    float* ho    = allocf((size_t)B * 243 * 69 * 69);
    float* loc   = allocf((size_t)B * NA_TOTAL * 4);
    float* conf  = allocf((size_t)B * NA_TOTAL * NCLS);
    float* coef  = allocf((size_t)B * NA_TOTAL * MASK_DIM);
    float* boxes = allocf((size_t)B * NA_TOTAL * 4);
    float* scr   = allocf((size_t)B * NA_TOTAL);
    float* ctop  = allocf((size_t)B * TOPK * MASK_DIM);
    int*   tidx  = (int*)(ws + off); off += ((B * TOPK * sizeof(int) + 255) / 256) * 256;

    float* out      = (float*)d_out;
    float* out_b    = out;           // [4,100,4]
    float* out_s    = out + 1600;    // [4,100]
    float* out_mask = out + 2000;    // [4,138,138,100]

    // ---- weight repack f32 -> f16 ----
    struct WC { const float* s; _Float16* d; int n; };
    const WC wcs[] = {
        {lat3_w, w16_lat3, F * 512},  {lat4_w, w16_lat4, F * 1024},
        {lat5_w, w16_lat5, F * 2048},
        {fpn3_w, w16_fpn3, F * K33},  {fpn4_w, w16_fpn4, F * K33},
        {fpn5_w, w16_fpn5, F * K33},  {ds1_w, w16_ds1, F * K33},
        {ds2_w, w16_ds2, F * K33},    {up_w, w16_up, F * K33},
        {bbox_w, w16_bbox, 12 * K33}, {conf_w, w16_conf, 243 * K33},
        {mask_w, w16_mask, 96 * K33},
        {pr1_w, w16_pr1, F * K33},    {pr2_w, w16_pr2, F * K33},
        {pr3_w, w16_pr3, F * K33},    {pr4_w, w16_pr4, F * K33},
        {pr5_w, w16_pr5, MASK_DIM * F},
    };
    for (const WC& wc : wcs)
        f32_to_f16<<<(wc.n + 255) / 256, 256, 0, stream>>>(wc.s, wc.d, wc.n);

    // ---- FPN ----
    launch_conv(stream, c5, w16_lat5, lat5_b, x5, B, 2048, 18, 18, F, 18, 18, 1, 0, 0, false);
    launch_conv(stream, c4, w16_lat4, lat4_b, x4, B, 1024, 35, 35, F, 35, 35, 1, 0, 0, false);
    {
        int tot = B * F * 35 * 35;
        upsample_bilinear<<<(tot + 255) / 256, 256, 0, stream>>>(x5, x4, B * F, 18, 18, 35, 35, 1);
    }
    launch_conv(stream, c3, w16_lat3, lat3_b, x3, B, 512, 69, 69, F, 69, 69, 1, 0, 0, false);
    {
        int tot = B * F * 69 * 69;
        upsample_bilinear<<<(tot + 255) / 256, 256, 0, stream>>>(x4, x3, B * F, 35, 35, 69, 69, 1);
    }
    launch_conv(stream, x3, w16_fpn3, fpn3_b, p3, B, F, 69, 69, F, 69, 69, 1, 1, 1, true);
    launch_conv(stream, x4, w16_fpn4, fpn4_b, p4, B, F, 35, 35, F, 35, 35, 1, 1, 1, true);
    launch_conv(stream, x5, w16_fpn5, fpn5_b, p5, B, F, 18, 18, F, 18, 18, 1, 1, 1, true);
    launch_conv(stream, p5, w16_ds1, ds1_b, p6, B, F, 18, 18, F, 9, 9, 2, 1, 0, true);
    launch_conv(stream, p6, w16_ds2, ds2_b, p7, B, F, 9, 9, F, 5, 5, 2, 1, 0, true);

    // ---- protonet on P3 ----
    launch_conv(stream, p3, w16_pr1, pr1_b, qa, B, F, 69, 69, F, 69, 69, 1, 1, 1, true);
    launch_conv(stream, qa, w16_pr2, pr2_b, qb, B, F, 69, 69, F, 69, 69, 1, 1, 1, true);
    launch_conv(stream, qb, w16_pr3, pr3_b, qa, B, F, 69, 69, F, 69, 69, 1, 1, 1, true);
    {
        int tot = B * F * 138 * 138;
        upsample_bilinear<<<(tot + 255) / 256, 256, 0, stream>>>(qa, qb, B * F, 69, 69, 138, 138, 0);
    }
    launch_conv(stream, qb, w16_pr4, pr4_b, qa, B, F, 138, 138, F, 138, 138, 1, 1, 1, true);
    launch_conv(stream, qa, w16_pr5, pr5_b, proto, B, F, 138, 138, MASK_DIM, 138, 138, 1, 0, 1, false);

    // ---- shared prediction heads over the 5 pyramid levels ----
    const float* lv_p[5] = {p3, p4, p5, p6, p7};
    const int    lv_h[5] = {69, 35, 18, 9, 5};
    const int    lv_base[5] = {0, 14283, 17958, 18930, 19173};
    for (int L = 0; L < 5; ++L) {
        int H = lv_h[L];
        launch_conv(stream, lv_p[L], w16_up, up_b, th, B, F, H, H, F, H, H, 1, 1, 1, true);

        launch_conv(stream, th, w16_bbox, bbox_b, ho, B, F, H, H, 12, H, H, 1, 1, 0, true);
        { long long tot = 4LL * H * H * 3 * 4;
          scatter_head<<<(int)((tot + 255) / 256), 256, 0, stream>>>(ho, loc, H, H, 4, lv_base[L]); }

        launch_conv(stream, th, w16_conf, conf_b, ho, B, F, H, H, 243, H, H, 1, 1, 0, true);
        { long long tot = 4LL * H * H * 3 * NCLS;
          scatter_head<<<(int)((tot + 255) / 256), 256, 0, stream>>>(ho, conf, H, H, NCLS, lv_base[L]); }

        launch_conv(stream, th, w16_mask, mask_b, ho, B, F, H, H, 96, H, H, 1, 1, 2, true);
        { long long tot = 4LL * H * H * 3 * MASK_DIM;
          scatter_head<<<(int)((tot + 255) / 256), 256, 0, stream>>>(ho, coef, H, H, MASK_DIM, lv_base[L]); }
    }

    // ---- decode + scores ----
    {
        int tot = B * NA_TOTAL;
        decode_score<<<(tot + 255) / 256, 256, 0, stream>>>(loc, conf, boxes, scr);
    }

    // ---- top-k, gather, masks ----
    topk_kernel<<<B, 256, 0, stream>>>(scr, tidx, out_s);
    gather_top<<<(B * TOPK + 255) / 256, 256, 0, stream>>>(boxes, coef, tidx, out_b, ctop);
    {
        int HW = 138 * 138;
        dim3 grid((HW + 255) / 256, B);
        mask_kernel<<<grid, 256, 0, stream>>>(proto, ctop, out_mask, HW);
    }
}
